// GravEGNNConv_32246614459256
// MI455X (gfx1250) — matrix operand order repaired and checked by reading, not compiled, not run on hardware
//
#include <hip/hip_runtime.h>
#include <hip/hip_bf16.h>
#include <math.h>

// ---------------------------------------------------------------------------
// Types for CDNA5 WMMA
// ---------------------------------------------------------------------------
typedef __attribute__((ext_vector_type(16))) __bf16 v16bf;
typedef __attribute__((ext_vector_type(8)))  float  v8f;
typedef __attribute__((ext_vector_type(4)))  int    v4i;

union FragBF { v16bf v; v4i q[2]; };
union AccF   { v8f  v; float f[8]; };

#define NDIM 64
#define HDIM 128
#define EDIM 8
#define KP1  160   // MSG_IN=138 padded to multiple of 32

__device__ __forceinline__ unsigned short f2bf(float f) {
    __bf16 b = (__bf16)f;                       // native v_cvt path
    return __builtin_bit_cast(unsigned short, b);
}
__device__ __forceinline__ float bf2f(unsigned short b) {
    union { unsigned u; float f; } x; x.u = ((unsigned)b) << 16;
    return x.f;
}
// fast silu: x * rcp(1 + exp(-x)) -- v_exp_f32 + v_rcp_f32, no IEEE div fixup
__device__ __forceinline__ float silu(float x) {
    return x * __builtin_amdgcn_rcpf(1.0f + __expf(-x));
}

// A-fragment (16xK tile, row-major bf16 in LDS, stride in bf16 elements).
// ISA 16-bit A layout: lanes0-15 row=lane, k = kc+0..7 & kc+16..23;
//                      lanes16-31 row=lane-16, k = kc+8..15 & kc+24..31.
__device__ __forceinline__ v16bf load_a_frag(const unsigned short* base, int stride,
                                             int kc, int lane) {
    const v4i* p = (const v4i*)(base + (lane & 15) * stride + kc + ((lane >> 4) << 3));
    FragBF f;
    f.q[0] = p[0];      // 8 bf16: k .. k+7
    f.q[1] = p[2];      // 8 bf16: k+16 .. k+23
    return f.v;
}

// B-fragment from prepacked weights: tile = kt*nTiles + nt, 32 lanes x 16 bf16
// contiguous per lane (packed by pack_w below to match ISA B layout).
__device__ __forceinline__ v16bf load_b_frag(const unsigned short* w, int tileIdx, int lane) {
    const v4i* p = (const v4i*)(w + (size_t)tileIdx * 512 + lane * 16);
    FragBF f;
    f.q[0] = p[0];
    f.q[1] = p[1];
    return f.v;
}

// ---------------------------------------------------------------------------
// Weight repack: f32 row-major [K][NN]  ->  bf16 WMMA B-fragments, K padded.
// B layout: lane<16: elem i = W[kt*32+(i<8?i:i+8)   ][nt*16+lane]
//           lane>=16: elem i = W[kt*32+8+(i<8?i:i+8)][nt*16+lane-16]
// ---------------------------------------------------------------------------
__global__ void pack_w(const float* __restrict__ W, unsigned short* __restrict__ dst,
                       int K, int Kpad, int NN) {
    int nTiles = NN >> 4;
    int total  = (Kpad >> 5) * nTiles * 32;
    int tid = blockIdx.x * blockDim.x + threadIdx.x;
    if (tid >= total) return;
    int lane = tid & 31;
    int tile = tid >> 5;
    int kt = tile / nTiles;
    int nt = tile - kt * nTiles;
    int n = nt * 16 + (lane & 15);
    int kbase = kt * 32 + ((lane >> 4) ? 8 : 0);
    unsigned short* o = dst + (size_t)tile * 512 + lane * 16;
#pragma unroll
    for (int i = 0; i < 16; ++i) {
        int k = kbase + (i < 8 ? i : i + 8);
        float v = (k < K) ? W[(size_t)k * NN + n] : 0.0f;
        o[i] = f2bf(v);
    }
}

__global__ void zero_f(float* p, long long n) {
    long long i = (long long)blockIdx.x * blockDim.x + threadIdx.x;
    if (i < n) p[i] = 0.0f;
}

__global__ void init_xv(const float* __restrict__ x, const float* __restrict__ v,
                        float* __restrict__ ox, float* __restrict__ ov, int n3) {
    int i = blockIdx.x * blockDim.x + threadIdx.x;
    if (i < n3) { ox[i] = x[i]; ov[i] = v[i]; }
}

// ---------------------------------------------------------------------------
// Edge kernel: one wave32 per 16-edge tile, 4 waves per block.
// ---------------------------------------------------------------------------
__global__ __launch_bounds__(128, 1) void edge_kernel(
    const float* __restrict__ h, const float* __restrict__ x,
    const int*  __restrict__ ei, const float* __restrict__ eattr,
    const float* __restrict__ bm1, const float* __restrict__ bm2,
    const float* __restrict__ bc1, const float* __restrict__ Wc2, const float* __restrict__ bc2,
    const float* __restrict__ bv1, const float* __restrict__ Wv2, const float* __restrict__ bv2,
    const unsigned short* __restrict__ Wm1p, const unsigned short* __restrict__ Wm2p,
    const unsigned short* __restrict__ Wc1p, const unsigned short* __restrict__ Wv1p,
    float* __restrict__ aggr, float* __restrict__ out_x, float* __restrict__ out_v,
    int Ecnt) {

    __shared__ __align__(16) unsigned short s_in[4][16 * KP1];
    __shared__ __align__(16) unsigned short s_cur[4][16 * HDIM];
    __shared__ __align__(16) unsigned short s_msg[4][16 * HDIM];
    __shared__ int   s_row[4][16];
    __shared__ float s_rel[4][16][3];
    __shared__ float s_inv[4][16];
    __shared__ float s_part[4][32];

    const int wave = threadIdx.x >> 5;
    const int lane = threadIdx.x & 31;
    const int tile = blockIdx.x * 4 + wave;
    if (tile * 16 >= Ecnt) return;

    unsigned short* mi   = s_in[wave];
    unsigned short* cur  = s_cur[wave];
    unsigned short* msgb = s_msg[wave];

    const int e    = lane & 15;
    const int half = lane >> 4;
    const long long ge = (long long)tile * 16 + e;
    const int r0 = ei[ge];
    const int c0 = ei[(long long)Ecnt + ge];

    // ---- gather msg_in tile: [h[row](64) | h[col](64) | sq | dz | eattr(8) | pad] ----
    if (half == 0) {
        float d0 = x[r0 * 3 + 0] - x[c0 * 3 + 0];
        float d1 = x[r0 * 3 + 1] - x[c0 * 3 + 1];
        float d2 = x[r0 * 3 + 2] - x[c0 * 3 + 2];
        float sq = d0 * d0 + d1 * d1 + d2 * d2;
        s_row[wave][e] = r0;
        s_rel[wave][e][0] = d0; s_rel[wave][e][1] = d1; s_rel[wave][e][2] = d2;
        s_inv[wave][e] = 1.0f / (sq + 1e-8f);
        mi[e * KP1 + 128] = f2bf(sq);
        mi[e * KP1 + 129] = f2bf(d2);
#pragma unroll
        for (int j = 0; j < EDIM; ++j) mi[e * KP1 + 130 + j] = f2bf(eattr[ge * EDIM + j]);
#pragma unroll
        for (int j = 138; j < KP1; ++j) mi[e * KP1 + j] = 0;
        const float4* hp = (const float4*)(h + (size_t)r0 * NDIM);
#pragma unroll
        for (int j = 0; j < 16; ++j) {
            float4 f = hp[j]; int b = e * KP1 + j * 4;
            mi[b + 0] = f2bf(f.x); mi[b + 1] = f2bf(f.y);
            mi[b + 2] = f2bf(f.z); mi[b + 3] = f2bf(f.w);
        }
    } else {
        const float4* hp = (const float4*)(h + (size_t)c0 * NDIM);
#pragma unroll
        for (int j = 0; j < 16; ++j) {
            float4 f = hp[j]; int b = e * KP1 + 64 + j * 4;
            mi[b + 0] = f2bf(f.x); mi[b + 1] = f2bf(f.y);
            mi[b + 2] = f2bf(f.z); mi[b + 3] = f2bf(f.w);
        }
    }
    __builtin_amdgcn_wave_barrier();

    const int colq = lane & 15;
    const int moff = half << 3;

    // ---- layer 1: 16x160 @ 160x128 -> silu -> cur ----
    for (int nt = 0; nt < 8; ++nt) {
        float b = bm1[nt * 16 + colq];
        AccF acc;
#pragma unroll
        for (int r = 0; r < 8; ++r) acc.f[r] = b;
#pragma unroll
        for (int kt = 0; kt < 5; ++kt) {
            v16bf a  = load_a_frag(mi, KP1, kt * 32, lane);
            v16bf bw = load_b_frag(Wm1p, kt * 8 + nt, lane);
            acc.v = __builtin_amdgcn_wmma_f32_16x16x32_bf16(false, a, false, bw,
                                                            (short)0, acc.v, false, false);
        }
        int colb = nt * 16 + colq;
#pragma unroll
        for (int r = 0; r < 8; ++r) cur[(moff + r) * HDIM + colb] = f2bf(silu(acc.f[r]));
    }
    __builtin_amdgcn_wave_barrier();

    // ---- layer 2: cur @ Wm2 -> silu -> msg ; scatter-add into aggr ----
    for (int nt = 0; nt < 8; ++nt) {
        float b = bm2[nt * 16 + colq];
        AccF acc;
#pragma unroll
        for (int r = 0; r < 8; ++r) acc.f[r] = b;
#pragma unroll
        for (int kt = 0; kt < 4; ++kt) {
            v16bf a  = load_a_frag(cur, HDIM, kt * 32, lane);
            v16bf bw = load_b_frag(Wm2p, kt * 8 + nt, lane);
            acc.v = __builtin_amdgcn_wmma_f32_16x16x32_bf16(false, a, false, bw,
                                                            (short)0, acc.v, false, false);
        }
        int colb = nt * 16 + colq;
#pragma unroll
        for (int r = 0; r < 8; ++r) {
            float m = silu(acc.f[r]);
            int M = moff + r;
            msgb[M * HDIM + colb] = f2bf(m);
            atomicAdd(&aggr[(size_t)s_row[wave][M] * HDIM + colb], m);
        }
    }
    __builtin_amdgcn_wave_barrier();

    // ---- coordinate head: t = silu(msg @ Wc1 + bc1); xw = (t.Wc2 + bc2)*inv ----
    for (int nt = 0; nt < 8; ++nt) {
        float b = bc1[nt * 16 + colq];
        AccF acc;
#pragma unroll
        for (int r = 0; r < 8; ++r) acc.f[r] = b;
#pragma unroll
        for (int kt = 0; kt < 4; ++kt) {
            v16bf a  = load_a_frag(msgb, HDIM, kt * 32, lane);
            v16bf bw = load_b_frag(Wc1p, kt * 8 + nt, lane);
            acc.v = __builtin_amdgcn_wmma_f32_16x16x32_bf16(false, a, false, bw,
                                                            (short)0, acc.v, false, false);
        }
        int colb = nt * 16 + colq;
#pragma unroll
        for (int r = 0; r < 8; ++r) cur[(moff + r) * HDIM + colb] = f2bf(silu(acc.f[r]));
    }
    __builtin_amdgcn_wave_barrier();
    {   // split 128-dot across both half-waves: this lane does k in [half*64, half*64+64)
        float s = 0.0f;
        int base = e * HDIM + half * 64;
        for (int k = 0; k < 64; ++k) s += bf2f(cur[base + k]) * Wc2[half * 64 + k];
        s_part[wave][lane] = s;
    }
    __builtin_amdgcn_wave_barrier();
    if (lane < 16) {
        float s = s_part[wave][lane] + s_part[wave][lane + 16];
        float xw = (s + bc2[0]) * s_inv[wave][lane];
        int node = s_row[wave][lane];
#pragma unroll
        for (int d = 0; d < 3; ++d)
            atomicAdd(&out_x[(size_t)node * 3 + d], s_rel[wave][lane][d] * xw);
    }
    __builtin_amdgcn_wave_barrier();

    // ---- velocity head ----
    for (int nt = 0; nt < 8; ++nt) {
        float b = bv1[nt * 16 + colq];
        AccF acc;
#pragma unroll
        for (int r = 0; r < 8; ++r) acc.f[r] = b;
#pragma unroll
        for (int kt = 0; kt < 4; ++kt) {
            v16bf a  = load_a_frag(msgb, HDIM, kt * 32, lane);
            v16bf bw = load_b_frag(Wv1p, kt * 8 + nt, lane);
            acc.v = __builtin_amdgcn_wmma_f32_16x16x32_bf16(false, a, false, bw,
                                                            (short)0, acc.v, false, false);
        }
        int colb = nt * 16 + colq;
#pragma unroll
        for (int r = 0; r < 8; ++r) cur[(moff + r) * HDIM + colb] = f2bf(silu(acc.f[r]));
    }
    __builtin_amdgcn_wave_barrier();
    {
        float s = 0.0f;
        int base = e * HDIM + half * 64;
        for (int k = 0; k < 64; ++k) s += bf2f(cur[base + k]) * Wv2[half * 64 + k];
        s_part[wave][lane] = s;
    }
    __builtin_amdgcn_wave_barrier();
    if (lane < 16) {
        float s = s_part[wave][lane] + s_part[wave][lane + 16];
        float vw = (s + bv2[0]) * s_inv[wave][lane];
        int node = s_row[wave][lane];
#pragma unroll
        for (int d = 0; d < 3; ++d)
            atomicAdd(&out_v[(size_t)node * 3 + d], s_rel[wave][lane][d] * vw);
    }
}

// ---------------------------------------------------------------------------
// Node kernel: h_new = h + silu([h|aggr] @ Wn1 + bn1) @ Wn2 + bn2
// ---------------------------------------------------------------------------
__global__ __launch_bounds__(128, 1) void node_kernel(
    const float* __restrict__ h, const float* __restrict__ aggr,
    const float* __restrict__ bn1, const float* __restrict__ bn2,
    const unsigned short* __restrict__ Wn1p, const unsigned short* __restrict__ Wn2p,
    float* __restrict__ out_h, int Ntot) {

    __shared__ __align__(16) unsigned short s_nin[4][16 * 192];
    __shared__ __align__(16) unsigned short s_c[4][16 * HDIM];

    const int wave = threadIdx.x >> 5;
    const int lane = threadIdx.x & 31;
    const int tile = blockIdx.x * 4 + wave;
    if (tile * 16 >= Ntot) return;

    unsigned short* nin = s_nin[wave];
    unsigned short* cur = s_c[wave];

    const int e    = lane & 15;
    const int half = lane >> 4;
    const int node = tile * 16 + e;

    if (half == 0) {
        const float4* hp = (const float4*)(h + (size_t)node * NDIM);
#pragma unroll
        for (int j = 0; j < 16; ++j) {
            float4 f = hp[j]; int b = e * 192 + j * 4;
            nin[b + 0] = f2bf(f.x); nin[b + 1] = f2bf(f.y);
            nin[b + 2] = f2bf(f.z); nin[b + 3] = f2bf(f.w);
        }
    } else {
        const float4* ap = (const float4*)(aggr + (size_t)node * HDIM);
#pragma unroll
        for (int j = 0; j < 32; ++j) {
            float4 f = ap[j]; int b = e * 192 + 64 + j * 4;
            nin[b + 0] = f2bf(f.x); nin[b + 1] = f2bf(f.y);
            nin[b + 2] = f2bf(f.z); nin[b + 3] = f2bf(f.w);
        }
    }
    __builtin_amdgcn_wave_barrier();

    const int colq = lane & 15;
    const int moff = half << 3;

    for (int nt = 0; nt < 8; ++nt) {
        float b = bn1[nt * 16 + colq];
        AccF acc;
#pragma unroll
        for (int r = 0; r < 8; ++r) acc.f[r] = b;
#pragma unroll
        for (int kt = 0; kt < 6; ++kt) {
            v16bf a  = load_a_frag(nin, 192, kt * 32, lane);
            v16bf bw = load_b_frag(Wn1p, kt * 8 + nt, lane);
            acc.v = __builtin_amdgcn_wmma_f32_16x16x32_bf16(false, a, false, bw,
                                                            (short)0, acc.v, false, false);
        }
        int colb = nt * 16 + colq;
#pragma unroll
        for (int r = 0; r < 8; ++r) cur[(moff + r) * HDIM + colb] = f2bf(silu(acc.f[r]));
    }
    __builtin_amdgcn_wave_barrier();

    for (int nt = 0; nt < 4; ++nt) {
        float b = bn2[nt * 16 + colq];
        AccF acc;
#pragma unroll
        for (int r = 0; r < 8; ++r) acc.f[r] = b;
#pragma unroll
        for (int kt = 0; kt < 4; ++kt) {
            v16bf a  = load_a_frag(cur, HDIM, kt * 32, lane);
            v16bf bw = load_b_frag(Wn2p, kt * 4 + nt, lane);
            acc.v = __builtin_amdgcn_wmma_f32_16x16x32_bf16(false, a, false, bw,
                                                            (short)0, acc.v, false, false);
        }
        int colb = nt * 16 + colq;
#pragma unroll
        for (int r = 0; r < 8; ++r) {
            int nn = tile * 16 + moff + r;
            out_h[(size_t)nn * NDIM + colb] = h[(size_t)nn * NDIM + colb] + acc.f[r];
        }
    }
}

// ---------------------------------------------------------------------------
// Launch
// ---------------------------------------------------------------------------
extern "C" void kernel_launch(void* const* d_in, const int* in_sizes, int n_in,
                              void* d_out, int out_size, void* d_ws, size_t ws_size,
                              hipStream_t stream) {
    const float* h    = (const float*)d_in[0];
    const float* x    = (const float*)d_in[1];
    const float* v    = (const float*)d_in[2];
    const int*   ei   = (const int*)d_in[3];
    const float* ea   = (const float*)d_in[4];
    const float* Wm1  = (const float*)d_in[5];
    const float* bm1  = (const float*)d_in[6];
    const float* Wm2  = (const float*)d_in[7];
    const float* bm2  = (const float*)d_in[8];
    const float* Wn1  = (const float*)d_in[9];
    const float* bn1  = (const float*)d_in[10];
    const float* Wn2  = (const float*)d_in[11];
    const float* bn2  = (const float*)d_in[12];
    const float* Wc1  = (const float*)d_in[13];
    const float* bc1  = (const float*)d_in[14];
    const float* Wc2  = (const float*)d_in[15];
    const float* bc2  = (const float*)d_in[16];
    const float* Wv1  = (const float*)d_in[17];
    const float* bv1  = (const float*)d_in[18];
    const float* Wv2  = (const float*)d_in[19];
    const float* bv2  = (const float*)d_in[20];

    const int N = in_sizes[0] / NDIM;
    const int E = in_sizes[3] / 2;

    // workspace layout (bf16 elements)
    unsigned short* us = (unsigned short*)d_ws;
    unsigned short* Wm1p = us;                 // 40 tiles * 512
    unsigned short* Wm2p = Wm1p + 40 * 512;    // 32 tiles
    unsigned short* Wc1p = Wm2p + 32 * 512;
    unsigned short* Wv1p = Wc1p + 32 * 512;
    unsigned short* Wn1p = Wv1p + 32 * 512;    // 48 tiles
    unsigned short* Wn2p = Wn1p + 48 * 512;    // 16 tiles
    float* aggr = (float*)(Wn2p + 16 * 512);   // N*128 f32

    float* out_h = (float*)d_out;
    float* out_x = out_h + (size_t)N * NDIM;
    float* out_v = out_x + (size_t)N * 3;

    // 1) repack weights to bf16 WMMA fragments
    pack_w<<<(40 * 32 + 255) / 256, 256, 0, stream>>>(Wm1, Wm1p, 138, 160, 128);
    pack_w<<<(32 * 32 + 255) / 256, 256, 0, stream>>>(Wm2, Wm2p, 128, 128, 128);
    pack_w<<<(32 * 32 + 255) / 256, 256, 0, stream>>>(Wc1, Wc1p, 128, 128, 128);
    pack_w<<<(32 * 32 + 255) / 256, 256, 0, stream>>>(Wv1, Wv1p, 128, 128, 128);
    pack_w<<<(48 * 32 + 255) / 256, 256, 0, stream>>>(Wn1, Wn1p, 192, 192, 128);
    pack_w<<<(16 * 32 + 255) / 256, 256, 0, stream>>>(Wn2, Wn2p, 128, 128, 64);

    // 2) zero aggregation buffer, seed x/v outputs
    long long na = (long long)N * HDIM;
    zero_f<<<(unsigned)((na + 255) / 256), 256, 0, stream>>>(aggr, na);
    init_xv<<<(N * 3 + 255) / 256, 256, 0, stream>>>(x, v, out_x, out_v, N * 3);

    // 3) edge pipeline (wave per 16-edge tile, 4 waves/block)
    int etiles = (E + 15) / 16;
    edge_kernel<<<(etiles + 3) / 4, 128, 0, stream>>>(
        h, x, ei, ea, bm1, bm2, bc1, Wc2, bc2, bv1, Wv2, bv2,
        Wm1p, Wm2p, Wc1p, Wv1p, aggr, out_x, out_v, E);

    // 4) node update
    int ntiles = (N + 15) / 16;
    node_kernel<<<(ntiles + 3) / 4, 128, 0, stream>>>(
        h, aggr, bn1, bn2, Wn1p, Wn2p, out_h, N);
}
